// MultiHeadAttention_65730179498463
// MI455X (gfx1250) — matrix operand order repaired and checked by reading, not compiled, not run on hardware
//
#include <hip/hip_runtime.h>

// ---------------------------------------------------------------------------
// MHA forward for gfx1250 (MI455X): bf16 WMMA everywhere, f32 softmax.
// B=2, S=2048, D=1024, H=16, Dh=64.
// GEMMs: async-to-LDS double-buffered tiles (ASYNCcnt) feeding v_wmma.
// ---------------------------------------------------------------------------

typedef __bf16 bf16_t;
typedef bf16_t v16bf __attribute__((ext_vector_type(16)));
typedef float  v8f   __attribute__((ext_vector_type(8)));
typedef int    v4i   __attribute__((ext_vector_type(4)));

union FragU { v16bf v; uint4 q[2]; };

__device__ __forceinline__ v8f v8f_zero() {
  v8f z = {0.f, 0.f, 0.f, 0.f, 0.f, 0.f, 0.f, 0.f};
  return z;
}

__device__ __forceinline__ unsigned short f2bf(float x) {
  unsigned int u = __float_as_uint(x);
  unsigned int r = (u + 0x7FFFu + ((u >> 16) & 1u)) >> 16;  // RNE
  return (unsigned short)r;
}

__device__ __forceinline__ v8f wmma_bf16(v16bf a, v16bf b, v8f c) {
  return __builtin_amdgcn_wmma_f32_16x16x32_bf16(
      false, a, false, b, (short)0, c, false, false);
}

// A fragment (16x32 bf16). rowptr = &A[row][k0]; elems 0..7 -> K=half*8+0..7,
// elems 8..15 -> K=16+half*8+0..7.
__device__ __forceinline__ v16bf load_frag_a(const unsigned short* rowptr, int half) {
  FragU f;
  const uint4* p = (const uint4*)(rowptr + half * 8);
  f.q[0] = p[0];
  f.q[1] = p[2];
  return f.v;
}

// B fragment (32x16 bf16). rowptr = &Bsrc[n][k0] (Bsrc row-major [N][K]).
// elems 0..7 -> K=half*8+0..7, elems 8..15 -> K=16+half*8+0..7.
__device__ __forceinline__ v16bf load_frag_b(const unsigned short* rowptr, int half) {
  FragU f;
  const uint4* p = (const uint4*)(rowptr + half * 8);
  f.q[0] = p[0];
  f.q[1] = p[2];
  return f.v;
}

// ---------------------------------------------------------------------------
// Async global->LDS copy (ASYNCcnt path) with synchronous fallback.
// Builtin signature (from hipcc diagnostic): (v4i as1*, v4i as3*, Ii, Ii)
// ---------------------------------------------------------------------------
#if __has_builtin(__builtin_amdgcn_global_load_async_to_lds_b128)
#define HAS_ASYNC_LDS 1
#endif

__device__ __forceinline__ void copy16_to_lds(const unsigned short* g, unsigned short* l) {
#if defined(HAS_ASYNC_LDS)
  v4i* gv = (v4i*)g;   // drop const, retype (generic)
  v4i* lv = (v4i*)l;
  __builtin_amdgcn_global_load_async_to_lds_b128(
      (__attribute__((address_space(1))) v4i*)gv,
      (__attribute__((address_space(3))) v4i*)lv, 0, 0);
#else
  *(uint4*)l = *(const uint4*)g;
#endif
}

__device__ __forceinline__ void wait_async() {
#if defined(HAS_ASYNC_LDS)
#if __has_builtin(__builtin_amdgcn_s_wait_asynccnt)
  __builtin_amdgcn_s_wait_asynccnt(0);
#else
  asm volatile("s_wait_asynccnt 0" ::: "memory");
#endif
#endif
}

// ---------------------------------------------------------------------------
// fp32 -> bf16 conversion (vectorized x4)
// ---------------------------------------------------------------------------
__global__ __launch_bounds__(256) void cvt_f32_bf16_v4(
    const float4* __restrict__ src, unsigned short* __restrict__ dst, int n4) {
  int i = blockIdx.x * blockDim.x + threadIdx.x;
  if (i < n4) {
    float4 v = src[i];
    uint2 o;
    o.x = (unsigned int)f2bf(v.x) | ((unsigned int)f2bf(v.y) << 16);
    o.y = (unsigned int)f2bf(v.z) | ((unsigned int)f2bf(v.w) << 16);
    *(uint2*)(dst + (size_t)i * 4) = o;
  }
}

// ---------------------------------------------------------------------------
// GEMM: C[m][n] = sum_k A[m][k] * W[n][k]   (M=4096, N=1024, K=1024)
// Block tile 128x128, 8 waves (4M x 2N), wave tile 32x64.
// A/B 128x32 tiles double-buffered in LDS, filled with async-to-LDS b128.
// mode 1: bf16 out head-split [b][h][s][d]
// mode 2: bf16 out head-split-T [b][h][d][s]   (V^T)
// mode 3: f32 out + bias, row-major            (final projection)
// ---------------------------------------------------------------------------
#define GK   1024
#define BM   128
#define BN   128
#define BK   32
#define LDA  40   // padded row stride (elems): 80B = 20 banks, conflict-free

__global__ __launch_bounds__(256) void gemm_wmma_bf16(
    const unsigned short* __restrict__ A, const unsigned short* __restrict__ W,
    unsigned short* __restrict__ outb, float* __restrict__ outf,
    const float* __restrict__ bias, int mode) {
  __shared__ unsigned short Alds[2][BM * LDA];   // 20 KB
  __shared__ unsigned short Blds[2][BN * LDA];   // 20 KB

  const int tid  = threadIdx.x;
  const int lane = tid & 31, wave = tid >> 5;
  const int wm = wave >> 1, wn = wave & 1;       // 4 x 2 waves
  const int l15 = lane & 15, half = lane >> 4;
  const int mblk = blockIdx.x * BM;
  const int nblk = blockIdx.y * BN;

  // Tile copy: BM*BK/8 = 512 16B-chunks per tile, 2 per thread per tile.
  // chunk c: row = c>>2, cc = c&3 (4 chunks per 32-elem row).
  auto issueTile = [&](int k0, int buf) {
    #pragma unroll
    for (int j = 0; j < 2; j++) {
      int c = tid + j * 256;
      int row = c >> 2, cc = c & 3;
      copy16_to_lds(A + (size_t)(mblk + row) * GK + k0 + cc * 8,
                    &Alds[buf][row * LDA + cc * 8]);
    }
    #pragma unroll
    for (int j = 0; j < 2; j++) {
      int c = tid + j * 256;
      int row = c >> 2, cc = c & 3;
      copy16_to_lds(W + (size_t)(nblk + row) * GK + k0 + cc * 8,
                    &Blds[buf][row * LDA + cc * 8]);
    }
  };

  v8f acc[2][4];
  #pragma unroll
  for (int mt = 0; mt < 2; mt++)
    #pragma unroll
    for (int nt = 0; nt < 4; nt++) acc[mt][nt] = v8f_zero();

  issueTile(0, 0);

  for (int k0 = 0; k0 < GK; k0 += BK) {
    const int buf = (k0 / BK) & 1;
    wait_async();        // my async writes for this tile landed
    __syncthreads();     // everyone's landed; prev buffer reads finished
    if (k0 + BK < GK) issueTile(k0 + BK, buf ^ 1);

    const unsigned short* Ab = &Alds[buf][(wm * 32) * LDA];
    const unsigned short* Bb = &Blds[buf][(wn * 64) * LDA];

    v16bf af[2];
    #pragma unroll
    for (int mt = 0; mt < 2; mt++)
      af[mt] = load_frag_a(Ab + (size_t)(mt * 16 + l15) * LDA, half);
    v16bf bf[4];
    #pragma unroll
    for (int nt = 0; nt < 4; nt++)
      bf[nt] = load_frag_b(Bb + (size_t)(nt * 16 + l15) * LDA, half);

    #pragma unroll
    for (int mt = 0; mt < 2; mt++)
      #pragma unroll
      for (int nt = 0; nt < 4; nt++)
        acc[mt][nt] = wmma_bf16(af[mt], bf[nt], acc[mt][nt]);
  }

  // Epilogue. C layout: VGPR r, lane l -> M = r + 8*(l>>4), N = l&15.
  #pragma unroll
  for (int mt = 0; mt < 2; mt++) {
    #pragma unroll
    for (int nt = 0; nt < 4; nt++) {
      #pragma unroll
      for (int r = 0; r < 8; r++) {
        int m = mblk + wm * 32 + mt * 16 + r + 8 * half;
        int n = nblk + wn * 64 + nt * 16 + l15;
        float val = acc[mt][nt][r];
        if (mode == 3) {
          outf[(size_t)m * 1024 + n] = val + bias[n];
        } else {
          int b = m >> 11, s = m & 2047;   // S = 2048
          int h = n >> 6,  d = n & 63;     // Dh = 64
          size_t idx = (mode == 1)
              ? ((size_t)(b * 16 + h) * 2048 + s) * 64 + d
              : ((size_t)(b * 16 + h) * 64 + d) * 2048 + s;
          outb[idx] = f2bf(val);
        }
      }
    }
  }
}

// ---------------------------------------------------------------------------
// Flash attention: one wave = 16 query rows of one (b,h).
// Qh, Kh: [B*H][S][Dh] bf16.  Vt: [B*H][Dh][S] bf16.  AO: [B*S][D] bf16.
// ---------------------------------------------------------------------------
__global__ __launch_bounds__(128) void attn_flash_wmma(
    const unsigned short* __restrict__ Qh, const unsigned short* __restrict__ Kh,
    const unsigned short* __restrict__ Vt, unsigned short* __restrict__ AO) {
  const int S = 2048, Dh = 64;
  const int lane = threadIdx.x & 31;
  const int wave = threadIdx.x >> 5;
  const int l15 = lane & 15, half = lane >> 4;
  const int bh = blockIdx.y;
  const int q0 = (blockIdx.x * 4 + wave) * 16;

  const unsigned short* Qp = Qh + (size_t)bh * S * Dh;
  const unsigned short* Kp = Kh + (size_t)bh * S * Dh;
  const unsigned short* Vp = Vt + (size_t)bh * Dh * S;

  __shared__ unsigned short Plds[4][16][32];  // per-wave P bounce buffer

  v16bf aq0 = load_frag_a(Qp + (size_t)(q0 + l15) * Dh + 0,  half);
  v16bf aq1 = load_frag_a(Qp + (size_t)(q0 + l15) * Dh + 32, half);

  v8f o[4];
  o[0] = v8f_zero(); o[1] = v8f_zero(); o[2] = v8f_zero(); o[3] = v8f_zero();
  float mrow[8], lrow[8];
  #pragma unroll
  for (int r = 0; r < 8; r++) { mrow[r] = -1e30f; lrow[r] = 0.f; }

  for (int kc = 0; kc < S; kc += 32) {
    v8f st[2];
    #pragma unroll
    for (int t = 0; t < 2; t++) {
      const unsigned short* krow = Kp + (size_t)(kc + t * 16 + l15) * Dh;
      v8f s = v8f_zero();
      s = wmma_bf16(aq0, load_frag_b(krow + 0,  half), s);
      s = wmma_bf16(aq1, load_frag_b(krow + 32, half), s);
      st[t] = s;
    }

    float alpha[8];
    #pragma unroll
    for (int r = 0; r < 8; r++) {
      float s0 = st[0][r] * 0.125f;   // 1/sqrt(64)
      float s1 = st[1][r] * 0.125f;
      float mc = fmaxf(s0, s1);
      mc = fmaxf(mc, __shfl_xor(mc, 1, 32));
      mc = fmaxf(mc, __shfl_xor(mc, 2, 32));
      mc = fmaxf(mc, __shfl_xor(mc, 4, 32));
      mc = fmaxf(mc, __shfl_xor(mc, 8, 32));
      float mn = fmaxf(mrow[r], mc);
      float p0 = __expf(s0 - mn);
      float p1 = __expf(s1 - mn);
      Plds[wave][r + 8 * half][l15]      = f2bf(p0);
      Plds[wave][r + 8 * half][16 + l15] = f2bf(p1);
      float ps = p0 + p1;
      ps += __shfl_xor(ps, 1, 32);
      ps += __shfl_xor(ps, 2, 32);
      ps += __shfl_xor(ps, 4, 32);
      ps += __shfl_xor(ps, 8, 32);
      float al = __expf(mrow[r] - mn);
      lrow[r] = lrow[r] * al + ps;
      mrow[r] = mn;
      alpha[r] = al;
    }
    #pragma unroll
    for (int r = 0; r < 8; r++) {
      o[0][r] *= alpha[r]; o[1][r] *= alpha[r];
      o[2][r] *= alpha[r]; o[3][r] *= alpha[r];
    }

    FragU pf;
    const uint4* pl = (const uint4*)(&Plds[wave][l15][half * 8]);
    pf.q[0] = pl[0];
    pf.q[1] = pl[2];

    #pragma unroll
    for (int nt = 0; nt < 4; nt++) {
      const unsigned short* vrow = Vp + (size_t)(nt * 16 + l15) * S + kc;
      o[nt] = wmma_bf16(pf.v, load_frag_b(vrow, half), o[nt]);
    }
  }

  const int b = bh >> 4, h = bh & 15;
  float inv[8];
  #pragma unroll
  for (int r = 0; r < 8; r++) inv[r] = 1.0f / lrow[r];
  #pragma unroll
  for (int nt = 0; nt < 4; nt++) {
    #pragma unroll
    for (int r = 0; r < 8; r++) {
      int s = q0 + r + 8 * half;
      size_t m = (size_t)b * S + s;
      AO[m * 1024 + h * 64 + nt * 16 + l15] = f2bf(o[nt][r] * inv[r]);
    }
  }
}

// ---------------------------------------------------------------------------
// Host-side orchestration
// ---------------------------------------------------------------------------
extern "C" void kernel_launch(void* const* d_in, const int* in_sizes, int n_in,
                              void* d_out, int out_size, void* d_ws, size_t ws_size,
                              hipStream_t stream) {
  (void)in_sizes; (void)n_in; (void)out_size; (void)ws_size;

  const float* q  = (const float*)d_in[0];
  const float* k  = (const float*)d_in[1];
  const float* v  = (const float*)d_in[2];
  const float* Wq = (const float*)d_in[3];
  const float* Wk = (const float*)d_in[4];
  const float* Wv = (const float*)d_in[5];
  const float* Wo = (const float*)d_in[6];
  const float* bo = (const float*)d_in[7];

  const size_t MEG = 1024ull * 1024ull;     // elements
  unsigned short* ws = (unsigned short*)d_ws;
  unsigned short* qb  = ws;                 // 4M bf16
  unsigned short* kb  = ws + 4  * MEG;
  unsigned short* vb  = ws + 8  * MEG;
  unsigned short* wqb = ws + 12 * MEG;
  unsigned short* wkb = ws + 13 * MEG;
  unsigned short* wvb = ws + 14 * MEG;
  unsigned short* wob = ws + 15 * MEG;
  unsigned short* Qh  = ws + 16 * MEG;      // [B*H][S][Dh]
  unsigned short* Kh  = ws + 20 * MEG;
  unsigned short* Vt  = ws + 24 * MEG;      // [B*H][Dh][S]
  unsigned short* AO  = ws + 28 * MEG;      // [B*S][D]

  auto cvt = [&](const float* s, unsigned short* d, size_t n) {
    int n4 = (int)(n / 4);
    cvt_f32_bf16_v4<<<(n4 + 255) / 256, 256, 0, stream>>>((const float4*)s, d, n4);
  };
  cvt(q,  qb,  4 * MEG);
  cvt(k,  kb,  4 * MEG);
  cvt(v,  vb,  4 * MEG);
  cvt(Wq, wqb, MEG);
  cvt(Wk, wkb, MEG);
  cvt(Wv, wvb, MEG);
  cvt(Wo, wob, MEG);

  dim3 gg(4096 / BM, 1024 / BN);
  gemm_wmma_bf16<<<gg, 256, 0, stream>>>(qb, wqb, Qh, nullptr, nullptr, 1);
  gemm_wmma_bf16<<<gg, 256, 0, stream>>>(kb, wkb, Kh, nullptr, nullptr, 1);
  gemm_wmma_bf16<<<gg, 256, 0, stream>>>(vb, wvb, Vt, nullptr, nullptr, 2);

  attn_flash_wmma<<<dim3(2048 / 64, 32), 128, 0, stream>>>(Qh, Kh, Vt, AO);

  gemm_wmma_bf16<<<gg, 256, 0, stream>>>(AO, wob, nullptr, (float*)d_out, bo, 3);
}
